// BertAdaSVDBlock_85529978733270
// MI455X (gfx1250) — compile-verified, matrix operand order checked by reading, not run on hardware
//
#include <hip/hip_runtime.h>
#include <hip/hip_bf16.h>
#include <math.h>

typedef __attribute__((ext_vector_type(16))) __bf16 bf16x16;
typedef __attribute__((ext_vector_type(8)))  __bf16 bf16x8;
typedef __attribute__((ext_vector_type(8)))  float  f32x8;

#define DMODEL 768
#define SEQ    2048
#define NBATCH 4
#define NHEAD  12
#define DHEAD  64
#define QKVLD  2304                 // 3 * DMODEL
#define ROWS   (NBATCH * SEQ)       // 8192

// ---------- bf16 helpers (bit-level; no native __bf16 arithmetic needed) ----------
__device__ __forceinline__ __bf16 f2bf(float f) {
  union { float f; unsigned u; } a; a.f = f;
  unsigned r = a.u + 0x7FFFu + ((a.u >> 16) & 1u);   // round-to-nearest-even
  union { unsigned short s; __bf16 b; } o; o.s = (unsigned short)(r >> 16);
  return o.b;
}
__device__ __forceinline__ float bf2f(__bf16 b) {
  union { __bf16 b; unsigned short s; } i; i.b = b;
  union { unsigned u; float f; } o; o.u = ((unsigned)i.s) << 16;
  return o.f;
}

// Load one 16x32 bf16 fragment (A layout; identical for B when p points at the
// transposed-B tile, i.e. row-major 16(cols) x 32(k)).
// Lane l: row = l&15, k-chunks at 8*(l>>4) and 8*(l>>4)+16 (each 8 contiguous bf16).
__device__ __forceinline__ bf16x16 load_frag(const __bf16* p, int ld) {
  int lane = threadIdx.x & 31;
  int r = lane & 15, hh = lane >> 4;
  const __bf16* q = p + (size_t)r * ld + 8 * hh;
  bf16x8 c0 = *(const bf16x8*)(q);
  bf16x8 c1 = *(const bf16x8*)(q + 16);
  return __builtin_shufflevector(c0, c1, 0,1,2,3,4,5,6,7,8,9,10,11,12,13,14,15);
}

__device__ __forceinline__ f32x8 wmma_bf16(bf16x16 a, bf16x16 b, f32x8 c) {
  return __builtin_amdgcn_wmma_f32_16x16x32_bf16(false, a, false, b, (short)0, c, false, false);
}

// ---------- fp32 -> bf16 (identity layout) ----------
__global__ void conv_bf16(const float* __restrict__ s, __bf16* __restrict__ d, long long n) {
  long long i = (long long)blockIdx.x * blockDim.x + threadIdx.x;
  if (i < n) d[i] = f2bf(s[i]);
}

// ---------- fp32 (K x N) -> bf16 transposed (N x K) ----------
__global__ void conv_bf16_t(const float* __restrict__ s, __bf16* __restrict__ d, int Kdim, int Ndim) {
  long long idx = (long long)blockIdx.x * blockDim.x + threadIdx.x;
  if (idx >= (long long)Kdim * Ndim) return;
  int k = (int)(idx % Kdim);
  int n = (int)(idx / Kdim);
  d[idx] = f2bf(s[(size_t)k * Ndim + n]);
}

// ---------- fold low-rank QKV weights into one transposed weight:
// WT[c][d] = sum_r P[h][d][r] * V[h][r][j],  c = p*768 + h*64 + j ----------
__global__ void build_wcat(const float* __restrict__ Pq, const float* __restrict__ Vq, const float* __restrict__ bq,
                           const float* __restrict__ Pk, const float* __restrict__ Vk, const float* __restrict__ bk,
                           const float* __restrict__ Pv, const float* __restrict__ Vv, const float* __restrict__ bv,
                           __bf16* __restrict__ WT, float* __restrict__ biascat) {
  long long idx = (long long)blockIdx.x * blockDim.x + threadIdx.x;
  if (idx >= (long long)DMODEL * QKVLD) return;
  int d = (int)(idx % DMODEL);
  int c = (int)(idx / DMODEL);
  int p   = c / DMODEL;
  int rem = c % DMODEL;
  int h = rem / DHEAD, j = rem % DHEAD;
  const float* P  = (p == 0) ? Pq : (p == 1) ? Pk : Pv;
  const float* V  = (p == 0) ? Vq : (p == 1) ? Vk : Vv;
  const float* bb = (p == 0) ? bq : (p == 1) ? bk : bv;
  const float* Pr = P + ((size_t)h * DMODEL + d) * 32;
  const float* Vc = V + (size_t)h * 32 * DHEAD + j;
  float acc = 0.f;
  for (int r = 0; r < 32; ++r) acc += Pr[r] * Vc[(size_t)r * DHEAD];
  WT[(size_t)c * DMODEL + d] = f2bf(acc);
  if (d == 0) biascat[c] = bb[h * DHEAD + j];
}

// ---------- extract V^T per (b,h): VT[bh][j][m] = QKV[b*SEQ+m][2*768 + h*64 + j] ----------
__global__ void transpose_v(const __bf16* __restrict__ QKV, __bf16* __restrict__ VT) {
  long long idx = (long long)blockIdx.x * blockDim.x + threadIdx.x;
  if (idx >= (long long)NBATCH * NHEAD * DHEAD * SEQ) return;
  int m  = (int)(idx % SEQ);
  long long rest = idx / SEQ;
  int j  = (int)(rest % DHEAD);
  int bh = (int)(rest / DHEAD);
  int b = bh / NHEAD, h = bh % NHEAD;
  VT[idx] = QKV[((size_t)(b * SEQ + m)) * QKVLD + 2 * DMODEL + h * DHEAD + j];
}

// ---------- bf16 WMMA GEMM: C(bf16) = act(A(MxK) @ W(KxN) + bias), W given transposed (NxK)
// 128 threads = 4 waves; 128x64 C tile; wave owns rows {m0+16w .. +16, m0+64+16w .. +16}.
// No LDS, no barriers: B fragments load directly from the pre-transposed weight. ----------
__global__ __launch_bounds__(128)
void gemm_bf16(const __bf16* __restrict__ A, const __bf16* __restrict__ WT,
               const float* __restrict__ bias, __bf16* __restrict__ C,
               int N, int K, int act) {
  int tid = threadIdx.x;
  int wave = tid >> 5, lane = tid & 31;
  int r = lane & 15, hh = lane >> 4;
  int n0 = blockIdx.x * 64;
  int m0 = blockIdx.y * 128;
  f32x8 acc[2][4];
  f32x8 z = {};
  for (int s = 0; s < 2; ++s)
    for (int i = 0; i < 4; ++i) acc[s][i] = z;
  const __bf16* A0 = A + (size_t)(m0 + wave * 16) * K;
  const __bf16* A1 = A0 + (size_t)64 * K;
  const __bf16* W0 = WT + (size_t)n0 * K;
  for (int kk = 0; kk < K; kk += 32) {
    __builtin_prefetch(A0 + kk + 128, 0, 1);
    __builtin_prefetch(W0 + kk + 128, 0, 1);
    bf16x16 af0 = load_frag(A0 + kk, K);
    bf16x16 af1 = load_frag(A1 + kk, K);
    bf16x16 b0 = load_frag(W0 + kk, K);
    bf16x16 b1 = load_frag(W0 + (size_t)16 * K + kk, K);
    bf16x16 b2 = load_frag(W0 + (size_t)32 * K + kk, K);
    bf16x16 b3 = load_frag(W0 + (size_t)48 * K + kk, K);
    acc[0][0] = wmma_bf16(af0, b0, acc[0][0]);
    acc[0][1] = wmma_bf16(af0, b1, acc[0][1]);
    acc[0][2] = wmma_bf16(af0, b2, acc[0][2]);
    acc[0][3] = wmma_bf16(af0, b3, acc[0][3]);
    acc[1][0] = wmma_bf16(af1, b0, acc[1][0]);
    acc[1][1] = wmma_bf16(af1, b1, acc[1][1]);
    acc[1][2] = wmma_bf16(af1, b2, acc[1][2]);
    acc[1][3] = wmma_bf16(af1, b3, acc[1][3]);
  }
  for (int s = 0; s < 2; ++s)
    for (int nf = 0; nf < 4; ++nf) {
      int col = n0 + nf * 16 + r;
      float bv = bias ? bias[col] : 0.0f;
      for (int i = 0; i < 8; ++i) {
        float v = acc[s][nf][i] + bv;
        if (act == 1) v = 0.5f * v * (1.0f + erff(v * 0.70710678118654752f));  // exact GELU
        int row = m0 + s * 64 + wave * 16 + i + 8 * hh;
        C[(size_t)row * N + col] = f2bf(v);
      }
    }
}

// ---------- flash attention: grid (SEQ/64, B*H), 128 threads, 4 independent waves,
// barrier-free (only LDS use is the wave-private P re-fragmentation tile). ----------
__global__ __launch_bounds__(128)
void flash_attn(const __bf16* __restrict__ QKV, const __bf16* __restrict__ VT,
                const float* __restrict__ mask, __bf16* __restrict__ ATTN) {
  __shared__ __align__(16) __bf16 ldsP[4][16 * 32];   // per-wave P tile (row-major 16x32)
  int tid = threadIdx.x, wave = tid >> 5, lane = tid & 31;
  int r = lane & 15, hh = lane >> 4;
  int bh = blockIdx.y, b = bh / NHEAD, h = bh % NHEAD;
  int m0 = blockIdx.x * 64;
  const __bf16* Qb  = QKV + ((size_t)(b * SEQ + m0 + wave * 16)) * QKVLD + h * DHEAD;
  const __bf16* Kb  = QKV + ((size_t)(b * SEQ)) * QKVLD + DMODEL + h * DHEAD;
  const __bf16* VTb = VT + (size_t)bh * DHEAD * SEQ;   // [64][2048]
  bf16x16 qf0 = load_frag(Qb, QKVLD);        // dh 0..31
  bf16x16 qf1 = load_frag(Qb + 32, QKVLD);   // dh 32..63
  f32x8 o[4]; f32x8 z = {};
  for (int i = 0; i < 4; ++i) o[i] = z;
  float mstat[8], sstat[8];
  for (int i = 0; i < 8; ++i) { mstat[i] = -3.0e38f; sstat[i] = 0.f; }
  __bf16* Pt = &ldsP[wave][0];

  for (int key0 = 0; key0 < SEQ; key0 += 32) {
    __builtin_prefetch(Kb + (size_t)(key0 + 32) * QKVLD, 0, 1);
    // S = Q K^T : row-major K (key x dh) is exactly the transposed-B layout.
    f32x8 s0 = z, s1 = z;
    {
      bf16x16 kf;
      kf = load_frag(Kb + (size_t)key0 * QKVLD, QKVLD);             s0 = wmma_bf16(qf0, kf, s0);
      kf = load_frag(Kb + (size_t)key0 * QKVLD + 32, QKVLD);        s0 = wmma_bf16(qf1, kf, s0);
      kf = load_frag(Kb + (size_t)(key0 + 16) * QKVLD, QKVLD);      s1 = wmma_bf16(qf0, kf, s1);
      kf = load_frag(Kb + (size_t)(key0 + 16) * QKVLD + 32, QKVLD); s1 = wmma_bf16(qf1, kf, s1);
    }
    float mk0 = mask[(size_t)b * SEQ + key0 + r];
    float mk1 = mask[(size_t)b * SEQ + key0 + 16 + r];
    for (int i = 0; i < 8; ++i) {   // this lane's row = i + 8*hh
      float a0 = s0[i] * 0.125f + mk0;   // 1/sqrt(64)
      float a1 = s1[i] * 0.125f + mk1;
      float rm = fmaxf(a0, a1);
      for (int w = 1; w < 16; w <<= 1) rm = fmaxf(rm, __shfl_xor(rm, w, 32));
      float nm = fmaxf(mstat[i], rm);
      float corr = __expf(mstat[i] - nm);
      mstat[i] = nm;
      float p0 = __expf(a0 - nm);
      float p1 = __expf(a1 - nm);
      float rs = p0 + p1;
      for (int w = 1; w < 16; w <<= 1) rs += __shfl_xor(rs, w, 32);
      sstat[i] = sstat[i] * corr + rs;
      o[0][i] *= corr; o[1][i] *= corr; o[2][i] *= corr; o[3][i] *= corr;
      int prow = i + 8 * hh;
      Pt[prow * 32 + r]      = f2bf(p0);
      Pt[prow * 32 + r + 16] = f2bf(p1);
    }
    // LDS ops are in-order within a wave (DScnt); tile is wave-private.
    bf16x16 pf = load_frag(Pt, 32);
    // O += P @ V : V^T rows (dh-major, contiguous keys) are the transposed-B layout.
    o[0] = wmma_bf16(pf, load_frag(VTb + (size_t)0  * SEQ + key0, SEQ), o[0]);
    o[1] = wmma_bf16(pf, load_frag(VTb + (size_t)16 * SEQ + key0, SEQ), o[1]);
    o[2] = wmma_bf16(pf, load_frag(VTb + (size_t)32 * SEQ + key0, SEQ), o[2]);
    o[3] = wmma_bf16(pf, load_frag(VTb + (size_t)48 * SEQ + key0, SEQ), o[3]);
  }
  for (int nf = 0; nf < 4; ++nf) {
    for (int i = 0; i < 8; ++i) {
      int qrow = m0 + wave * 16 + i + 8 * hh;
      float v = o[nf][i] / sstat[i];
      ATTN[((size_t)(b * SEQ + qrow)) * DMODEL + h * DHEAD + nf * 16 + r] = f2bf(v);
    }
  }
}

// ---------- fused residual + LayerNorm (fp32 math) ----------
__global__ __launch_bounds__(256)
void ln_kernel(const float* __restrict__ xf, const __bf16* __restrict__ xb,
               const __bf16* __restrict__ yb,
               const float* __restrict__ g, const float* __restrict__ beta,
               __bf16* __restrict__ outb, float* __restrict__ outf) {
  __shared__ float red[256];
  int row = blockIdx.x, t = threadIdx.x;
  size_t base = (size_t)row * DMODEL;
  float vals[3];
  for (int i = 0; i < 3; ++i) {
    int c = t + i * 256;
    float a = xf ? xf[base + c] : bf2f(xb[base + c]);
    vals[i] = a + bf2f(yb[base + c]);
  }
  red[t] = vals[0] + vals[1] + vals[2];
  __syncthreads();
  for (int off = 128; off > 0; off >>= 1) { if (t < off) red[t] += red[t + off]; __syncthreads(); }
  float mu = red[0] * (1.0f / DMODEL);
  __syncthreads();
  float vs = 0.f;
  for (int i = 0; i < 3; ++i) { float d = vals[i] - mu; vs += d * d; }
  red[t] = vs;
  __syncthreads();
  for (int off = 128; off > 0; off >>= 1) { if (t < off) red[t] += red[t + off]; __syncthreads(); }
  float inv = rsqrtf(red[0] * (1.0f / DMODEL) + 1e-12f);
  for (int i = 0; i < 3; ++i) {
    int c = t + i * 256;
    float ov = (vals[i] - mu) * inv * g[c] + beta[c];
    if (outb) outb[base + c] = f2bf(ov);
    if (outf) outf[base + c] = ov;
  }
}

extern "C" void kernel_launch(void* const* d_in, const int* in_sizes, int n_in,
                              void* d_out, int out_size, void* d_ws, size_t ws_size,
                              hipStream_t stream) {
  (void)in_sizes; (void)n_in; (void)out_size; (void)ws_size;
  const float* x   = (const float*)d_in[0];
  const float* msk = (const float*)d_in[1];
  const float* Pq  = (const float*)d_in[2];
  const float* Vq  = (const float*)d_in[3];
  const float* bq  = (const float*)d_in[4];
  const float* Pk  = (const float*)d_in[5];
  const float* Vk  = (const float*)d_in[6];
  const float* bk  = (const float*)d_in[7];
  const float* Pv  = (const float*)d_in[8];
  const float* Vv  = (const float*)d_in[9];
  const float* bv  = (const float*)d_in[10];
  const float* Uo  = (const float*)d_in[11];
  const float* Vo  = (const float*)d_in[12];
  const float* bo  = (const float*)d_in[13];
  const float* U1  = (const float*)d_in[14];
  const float* V1  = (const float*)d_in[15];
  const float* b1  = (const float*)d_in[16];
  const float* U2  = (const float*)d_in[17];
  const float* V2  = (const float*)d_in[18];
  const float* b2  = (const float*)d_in[19];
  const float* g1  = (const float*)d_in[20];
  const float* be1 = (const float*)d_in[21];
  const float* g2  = (const float*)d_in[22];
  const float* be2 = (const float*)d_in[23];

  char* ws = (char*)d_ws;
  size_t off = 0;
  auto take = [&](size_t bytes) { void* p = ws + off; off += (bytes + 255) & ~(size_t)255; return p; };
  __bf16* XB    = (__bf16*)take((size_t)ROWS * DMODEL * 2);
  __bf16* WCATT = (__bf16*)take((size_t)QKVLD * DMODEL * 2);    // [2304][768]
  float*  BCAT  = (float*) take((size_t)QKVLD * 4);
  __bf16* QKV   = (__bf16*)take((size_t)ROWS * QKVLD * 2);
  __bf16* VT    = (__bf16*)take((size_t)NBATCH * NHEAD * DHEAD * SEQ * 2);
  __bf16* ATTN  = (__bf16*)take((size_t)ROWS * DMODEL * 2);
  __bf16* T1    = (__bf16*)take((size_t)ROWS * 256 * 2);        // reused T1 / MID / T2
  __bf16* Y     = (__bf16*)take((size_t)ROWS * DMODEL * 2);     // reused Y1 / Y2
  __bf16* X1B   = (__bf16*)take((size_t)ROWS * DMODEL * 2);
  __bf16* ACT   = (__bf16*)take((size_t)ROWS * 3072 * 2);
  __bf16* UoT   = (__bf16*)take((size_t)256 * 768 * 2);         // [256][768]
  __bf16* VoT   = (__bf16*)take((size_t)768 * 256 * 2);         // [768][256]
  __bf16* U1T   = (__bf16*)take((size_t)256 * 768 * 2);
  __bf16* V1T   = (__bf16*)take((size_t)3072 * 256 * 2);
  __bf16* U2T   = (__bf16*)take((size_t)256 * 3072 * 2);
  __bf16* V2T   = (__bf16*)take((size_t)768 * 256 * 2);

  auto cdiv = [](long long a, long long b) { return (unsigned)((a + b - 1) / b); };

  // prep: x -> bf16; all weights -> bf16 TRANSPOSED (N x K); fold P@V for QKV
  conv_bf16<<<cdiv((long long)ROWS * DMODEL, 256), 256, 0, stream>>>(x, XB, (long long)ROWS * DMODEL);
  conv_bf16_t<<<cdiv(768 * 256, 256),  256, 0, stream>>>(Uo, UoT, 768, 256);
  conv_bf16_t<<<cdiv(256 * 768, 256),  256, 0, stream>>>(Vo, VoT, 256, 768);
  conv_bf16_t<<<cdiv(768 * 256, 256),  256, 0, stream>>>(U1, U1T, 768, 256);
  conv_bf16_t<<<cdiv(256 * 3072, 256), 256, 0, stream>>>(V1, V1T, 256, 3072);
  conv_bf16_t<<<cdiv(3072 * 256, 256), 256, 0, stream>>>(U2, U2T, 3072, 256);
  conv_bf16_t<<<cdiv(256 * 768, 256),  256, 0, stream>>>(V2, V2T, 256, 768);
  build_wcat<<<cdiv((long long)DMODEL * QKVLD, 256), 256, 0, stream>>>(
      Pq, Vq, bq, Pk, Vk, bk, Pv, Vv, bv, WCATT, BCAT);

  // fused QKV projection: [8192 x 2304] = XB @ Wcat + biascat
  gemm_bf16<<<dim3(QKVLD / 64, ROWS / 128), 128, 0, stream>>>(XB, WCATT, BCAT, QKV, QKVLD, DMODEL, 0);

  // V^T extraction, then attention
  transpose_v<<<cdiv((long long)NBATCH * NHEAD * DHEAD * SEQ, 256), 256, 0, stream>>>(QKV, VT);
  flash_attn<<<dim3(SEQ / 64, NBATCH * NHEAD), 128, 0, stream>>>(QKV, VT, msk, ATTN);

  // low-rank output projection + residual + LN1
  gemm_bf16<<<dim3(256 / 64, ROWS / 128),    128, 0, stream>>>(ATTN, UoT, nullptr, T1, 256, DMODEL, 0);
  gemm_bf16<<<dim3(DMODEL / 64, ROWS / 128), 128, 0, stream>>>(T1, VoT, bo, Y, DMODEL, 256, 0);
  ln_kernel<<<ROWS, 256, 0, stream>>>(x, nullptr, Y, g1, be1, X1B, nullptr);

  // low-rank FFN with exact GELU
  gemm_bf16<<<dim3(256 / 64, ROWS / 128),    128, 0, stream>>>(X1B, U1T, nullptr, T1, 256, DMODEL, 0);
  gemm_bf16<<<dim3(3072 / 64, ROWS / 128),   128, 0, stream>>>(T1, V1T, b1, ACT, 3072, 256, 1);
  gemm_bf16<<<dim3(256 / 64, ROWS / 128),    128, 0, stream>>>(ACT, U2T, nullptr, T1, 256, 3072, 0);
  gemm_bf16<<<dim3(DMODEL / 64, ROWS / 128), 128, 0, stream>>>(T1, V2T, b2, Y, DMODEL, 256, 0);

  // residual + LN2 -> fp32 output
  ln_kernel<<<ROWS, 256, 0, stream>>>(nullptr, X1B, Y, g2, be2, nullptr, (float*)d_out);
}
